// AttnDecoderLSTM_40398462386269
// MI455X (gfx1250) — matrix operand order repaired and checked by reading, not compiled, not run on hardware
//
#include <hip/hip_runtime.h>
#include <hip/hip_bf16.h>
#include <math.h>

typedef __attribute__((ext_vector_type(16))) _Float16 v16h;
typedef __attribute__((ext_vector_type(8)))  float    v8f;
typedef __attribute__((ext_vector_type(4)))  float    f4;

#define B_DIM 4096
#define S_DIM 512
#define H_DIM 128
#define O_DIM 128

// ---------------------------------------------------------------------------
// WMMA helper: D = A(16x32 f16) * B(32x16 f16) + C(16x16 f32)
// ---------------------------------------------------------------------------
__device__ __forceinline__ v8f wmma16(v16h a, v16h b, v8f c) {
  return __builtin_amdgcn_wmma_f32_16x16x32_f16(
      /*neg_a=*/false, a, /*neg_b=*/false, b,
      /*c_mod=*/(short)0, c, /*reuse_a=*/false, /*reuse_b=*/false);
}

// A-matrix fragment loader (ISA 16-bit A 16x32 layout):
//   lane L: m = L&15, kh = L>>4
//   halves 0..7  = A[m, kb*32 + 8*kh + e]
//   halves 8..15 = A[m, kb*32 + 16 + 8*kh + e]
// `row` points at element [m, 0] of a row-major matrix (fp32, global or LDS).
__device__ __forceinline__ v16h loadA(const float* row, int kb, int kh) {
  const float* p = row + kb * 32 + 8 * kh;
  v16h r;
#pragma unroll
  for (int e = 0; e < 8; ++e) r[e] = (_Float16)p[e];
#pragma unroll
  for (int e = 0; e < 8; ++e) r[8 + e] = (_Float16)p[16 + e];
  return r;
}

// B-matrix fragment loader from pre-packed f16 weights.
// Packed layout: group index g = (nt*KB + kb)*32 + lane, 16 halves per group
// (one aligned 32-byte load per lane).
__device__ __forceinline__ v16h loadB(const _Float16* __restrict__ Wp,
                                      int nt, int KB, int kb, int lane) {
  return *(const v16h*)(Wp + ((((size_t)nt * KB + kb) << 5) + lane) * 16);
}

__device__ __forceinline__ float sigm(float v) { return 1.0f / (1.0f + __expf(-v)); }

// ---------------------------------------------------------------------------
// Kernel 0: pack fp32 weight W (N x K, row-major, used as B in D = A @ W^T)
// into f16 WMMA B-layout: lane L (n = nt*16 + L&15, kh = L>>4) holds
// W[n, kb*32 + 8*kh + 0..7] then W[n, kb*32 + 16 + 8*kh + 0..7].
// ---------------------------------------------------------------------------
__global__ __launch_bounds__(256) void pack_b_kernel(
    const float* __restrict__ W, _Float16* __restrict__ Wp, int N, int K) {
  const int KB    = K >> 5;
  const int total = (N >> 4) * KB * 32;
  const int t = blockIdx.x * 256 + threadIdx.x;
  if (t >= total) return;
  const int L  = t & 31;
  const int kb = (t >> 5) % KB;
  const int nt = t / (KB << 5);
  const int n  = (nt << 4) + (L & 15);
  const int kh = L >> 4;
  const float* src = W + (size_t)n * K + (kb << 5) + (kh << 3);
  _Float16*    dst = Wp + ((size_t)t << 4);
#pragma unroll
  for (int e = 0; e < 16; ++e) dst[e] = (_Float16)src[e < 8 ? e : e + 8];
}

// ---------------------------------------------------------------------------
// Kernel 1: attention scores (WMMA GEMM 16x256 @ 256x512) + softmax.
// One wave (32 threads) per 16-row batch tile.  Scores staged in LDS,
// softmax done by lanes 0..15 (one row each), result written to d_out.
// ---------------------------------------------------------------------------
__global__ __launch_bounds__(32) void attn_softmax_kernel(
    const float* __restrict__ x, const float* __restrict__ h0,
    const _Float16* __restrict__ attn_p, const float* __restrict__ attn_b,
    float* __restrict__ attn_out) {
  __shared__ float ss[16 * S_DIM];  // 32 KB
  const int lane = threadIdx.x;
  const int m    = lane & 15;
  const int kh   = lane >> 4;
  const int row0 = blockIdx.x << 4;

  const float* xr = x  + (size_t)(row0 + m) * O_DIM;
  const float* hr = h0 + (size_t)(row0 + m) * H_DIM;
  v16h ax[4], ah[4];
#pragma unroll
  for (int kb = 0; kb < 4; ++kb) {
    ax[kb] = loadA(xr, kb, kh);
    ah[kb] = loadA(hr, kb, kh);
  }

  for (int nt = 0; nt < 32; ++nt) {               // 32 N-tiles over S=512
    v8f c = {};
#pragma unroll
    for (int kb = 0; kb < 4; ++kb) c = wmma16(ax[kb], loadB(attn_p, nt, 8, kb,     lane), c);
#pragma unroll
    for (int kb = 0; kb < 4; ++kb) c = wmma16(ah[kb], loadB(attn_p, nt, 8, kb + 4, lane), c);
    const float bb = attn_b[nt * 16 + m];
#pragma unroll
    for (int r = 0; r < 8; ++r)
      ss[(r + 8 * kh) * S_DIM + nt * 16 + m] = c[r] + bb;
  }

  if (lane < 16) {                                 // softmax, one row per lane
    float* rp = ss + m * S_DIM;
    float mx = rp[0];
    for (int s = 1; s < S_DIM; ++s) mx = fmaxf(mx, rp[s]);
    float sum = 0.0f;
    for (int s = 0; s < S_DIM; ++s) { float e = __expf(rp[s] - mx); rp[s] = e; sum += e; }
    const float inv = 1.0f / sum;
    float* gp = attn_out + (size_t)(row0 + m) * S_DIM;
#pragma unroll 4
    for (int s = 0; s < S_DIM; ++s) gp[s] = rp[s] * inv;
  }
}

// ---------------------------------------------------------------------------
// Kernel 2: ctx[b,h] = sum_s w[b,s] * enc[b,s,h]   (per-row B matrix -> VALU)
// This kernel carries the 1.07 GB HBM stream: 4 waves/block, one batch row
// per wave, float4 non-temporal loads, unrolled for deep LOADcnt.
// ---------------------------------------------------------------------------
__global__ __launch_bounds__(128) void ctx_kernel(
    const float* __restrict__ attn, const float* __restrict__ enc,
    float* __restrict__ ctx) {
  __shared__ float wbuf[4 * S_DIM];  // 8 KB
  const int b0 = blockIdx.x * 4;
  for (int i = threadIdx.x; i < 4 * S_DIM; i += 128)
    wbuf[i] = attn[(size_t)b0 * S_DIM + i];
  __syncthreads();

  const int wave = threadIdx.x >> 5;
  const int lane = threadIdx.x & 31;
  const int b    = b0 + wave;
  const int ch   = lane * 4;                        // 32 lanes x float4 = 128 ch
  const f4* ep = (const f4*)(enc + (size_t)b * S_DIM * H_DIM + ch);
  const float* wp = wbuf + wave * S_DIM;

  f4 acc = {};
#pragma unroll 8
  for (int s = 0; s < S_DIM; ++s) {
    f4 e = __builtin_nontemporal_load(ep + s * (H_DIM / 4));  // TH=NT: stream past L2
    acc += wp[s] * e;
  }
  *(f4*)(ctx + (size_t)b * H_DIM + ch) = acc;
}

// ---------------------------------------------------------------------------
// Kernel 3: comb -> gates -> LSTM cell -> out.  One wave per 16-row tile,
// all four GEMM stages on WMMA, LSTM elementwise in fp32 VALU.
// ---------------------------------------------------------------------------
__global__ __launch_bounds__(32) void lstm_tail_kernel(
    const float* __restrict__ x, const float* __restrict__ h0,
    const float* __restrict__ c0, const float* __restrict__ ctx,
    const _Float16* __restrict__ comb_p, const float* __restrict__ comb_b,
    const _Float16* __restrict__ wih_p,  const float* __restrict__ b_ih,
    const _Float16* __restrict__ whh_p,  const float* __restrict__ b_hh,
    const _Float16* __restrict__ outw_p, const float* __restrict__ out_b,
    float* __restrict__ out, float* __restrict__ h1o, float* __restrict__ c1o) {
  __shared__ float g[16 * 512];   // gates, 32 KB
  __shared__ float t2[16 * 128];  // comb, then h1, 8 KB
  const int lane = threadIdx.x;
  const int m    = lane & 15;
  const int kh   = lane >> 4;
  const int row0 = blockIdx.x << 4;

  const float* xr = x   + (size_t)(row0 + m) * O_DIM;
  const float* hr = h0  + (size_t)(row0 + m) * H_DIM;
  const float* cr = ctx + (size_t)(row0 + m) * H_DIM;
  v16h ax[4], ah[4], ac[4];
#pragma unroll
  for (int kb = 0; kb < 4; ++kb) {
    ax[kb] = loadA(xr, kb, kh);
    ah[kb] = loadA(hr, kb, kh);
    ac[kb] = loadA(cr, kb, kh);
  }

  // comb = relu([x | ctx] @ comb_W^T + comb_b)   (16x256 @ 256x128)
#pragma unroll
  for (int nt = 0; nt < 8; ++nt) {
    v8f c = {};
#pragma unroll
    for (int kb = 0; kb < 4; ++kb) c = wmma16(ax[kb], loadB(comb_p, nt, 8, kb,     lane), c);
#pragma unroll
    for (int kb = 0; kb < 4; ++kb) c = wmma16(ac[kb], loadB(comb_p, nt, 8, kb + 4, lane), c);
    const float bb = comb_b[nt * 16 + m];
#pragma unroll
    for (int r = 0; r < 8; ++r)
      t2[(r + 8 * kh) * 128 + nt * 16 + m] = fmaxf(c[r] + bb, 0.0f);
  }

  v16h acb[4];
  const float* cbr = t2 + m * 128;
#pragma unroll
  for (int kb = 0; kb < 4; ++kb) acb[kb] = loadA(cbr, kb, kh);

  // gates = comb @ W_ih^T + h0 @ W_hh^T + b_ih + b_hh   (16x128 @ 128x512, x2)
  for (int nt = 0; nt < 32; ++nt) {
    v8f c = {};
#pragma unroll
    for (int kb = 0; kb < 4; ++kb) c = wmma16(acb[kb], loadB(wih_p, nt, 4, kb, lane), c);
#pragma unroll
    for (int kb = 0; kb < 4; ++kb) c = wmma16(ah[kb],  loadB(whh_p, nt, 4, kb, lane), c);
    const float bb = b_ih[nt * 16 + m] + b_hh[nt * 16 + m];
#pragma unroll
    for (int r = 0; r < 8; ++r)
      g[(r + 8 * kh) * 512 + nt * 16 + m] = c[r] + bb;
  }

  // LSTM elementwise: lane owns 4 channels across all 16 rows
  const int ch = lane * 4;
  for (int mm = 0; mm < 16; ++mm) {
    const float* gr = g + mm * 512;
    f4 ig = *(const f4*)(gr + ch);
    f4 fg = *(const f4*)(gr + 128 + ch);
    f4 gg = *(const f4*)(gr + 256 + ch);
    f4 og = *(const f4*)(gr + 384 + ch);
    f4 cv = *(const f4*)(c0 + (size_t)(row0 + mm) * H_DIM + ch);
    f4 c1v, h1v;
#pragma unroll
    for (int e = 0; e < 4; ++e) {
      const float i_ = sigm(ig[e]);
      const float f_ = sigm(fg[e]);
      const float g_ = tanhf(gg[e]);
      const float o_ = sigm(og[e]);
      const float cc = f_ * cv[e] + i_ * g_;
      c1v[e] = cc;
      h1v[e] = o_ * tanhf(cc);
    }
    *(f4*)(c1o + (size_t)(row0 + mm) * H_DIM + ch) = c1v;
    *(f4*)(h1o + (size_t)(row0 + mm) * H_DIM + ch) = h1v;
    *(f4*)(t2 + mm * 128 + ch) = h1v;
  }

  // out = h1 @ out_W^T + out_b   (16x128 @ 128x128)
  v16h ah1[4];
  const float* h1r = t2 + m * 128;
#pragma unroll
  for (int kb = 0; kb < 4; ++kb) ah1[kb] = loadA(h1r, kb, kh);
#pragma unroll
  for (int nt = 0; nt < 8; ++nt) {
    v8f c = {};
#pragma unroll
    for (int kb = 0; kb < 4; ++kb) c = wmma16(ah1[kb], loadB(outw_p, nt, 4, kb, lane), c);
    const float bb = out_b[nt * 16 + m];
#pragma unroll
    for (int r = 0; r < 8; ++r)
      out[(size_t)(row0 + r + 8 * kh) * O_DIM + nt * 16 + m] = c[r] + bb;
  }
}

// ---------------------------------------------------------------------------
// Host launcher
// ---------------------------------------------------------------------------
extern "C" void kernel_launch(void* const* d_in, const int* in_sizes, int n_in,
                              void* d_out, int out_size, void* d_ws, size_t ws_size,
                              hipStream_t stream) {
  (void)in_sizes; (void)n_in; (void)out_size; (void)ws_size;

  const float* x      = (const float*)d_in[0];   // (B,1,O)
  const float* h0     = (const float*)d_in[1];   // (1,B,H)
  const float* c0     = (const float*)d_in[2];   // (1,B,H)
  const float* enc    = (const float*)d_in[3];   // (B,S,H)
  const float* attn_W = (const float*)d_in[4];   // (S, O+H) = (512,256)
  const float* attn_b = (const float*)d_in[5];   // (S,)
  const float* comb_W = (const float*)d_in[6];   // (O, O+H) = (128,256)
  const float* comb_b = (const float*)d_in[7];   // (O,)
  const float* W_ih   = (const float*)d_in[8];   // (512,128)
  const float* b_ih   = (const float*)d_in[9];   // (512,)
  const float* W_hh   = (const float*)d_in[10];  // (512,128)
  const float* b_hh   = (const float*)d_in[11];  // (512,)
  const float* out_W  = (const float*)d_in[12];  // (128,128)
  const float* out_b  = (const float*)d_in[13];  // (128,)

  // Workspace layout (f16 packed weights, then fp32 ctx)
  _Float16* wsh   = (_Float16*)d_ws;
  _Float16* attn_p = wsh;                 // 131072 halves
  _Float16* comb_p = wsh + 131072;        //  32768
  _Float16* wih_p  = wsh + 163840;        //  65536
  _Float16* whh_p  = wsh + 229376;        //  65536
  _Float16* outw_p = wsh + 294912;        //  16384
  float*    ctx_ws = (float*)(wsh + 311296);  // B*H floats (2 MB)

  // d_out layout: out | h1 | c1 | attn_weights (fp32, concatenated flat)
  float* out      = (float*)d_out;
  float* h1_out   = out + (size_t)B_DIM * O_DIM;      //  524288
  float* c1_out   = h1_out + (size_t)B_DIM * H_DIM;   // 1048576
  float* attn_out = c1_out + (size_t)B_DIM * H_DIM;   // 1572864

  // Pack weights to f16 WMMA B-layout (all tiny; L2-resident afterwards)
  pack_b_kernel<<<32, 256, 0, stream>>>(attn_W, attn_p, 512, 256);
  pack_b_kernel<<< 8, 256, 0, stream>>>(comb_W, comb_p, 128, 256);
  pack_b_kernel<<<16, 256, 0, stream>>>(W_ih,  wih_p,  512, 128);
  pack_b_kernel<<<16, 256, 0, stream>>>(W_hh,  whh_p,  512, 128);
  pack_b_kernel<<< 4, 256, 0, stream>>>(out_W, outw_p, 128, 128);

  // Attention scores + softmax (WMMA), writes attn_weights output
  attn_softmax_kernel<<<B_DIM / 16, 32, 0, stream>>>(x, h0, attn_p, attn_b, attn_out);

  // Context einsum: the 1.07 GB HBM stream (non-temporal)
  ctx_kernel<<<B_DIM / 4, 128, 0, stream>>>(attn_out, enc, ctx_ws);

  // comb/gates/LSTM/out (WMMA)
  lstm_tail_kernel<<<B_DIM / 16, 32, 0, stream>>>(
      x, h0, c0, ctx_ws,
      comb_p, comb_b, wih_p, b_ih, whh_p, b_hh, outw_p, out_b,
      out, h1_out, c1_out);
}
